// _NonLocalBlockND_18554258719373
// MI455X (gfx1250) — matrix operand order repaired and checked by reading, not compiled, not run on hardware
//
#include <hip/hip_runtime.h>
#include <hip/hip_bf16.h>

// ---------------------------------------------------------------------------
// NonLocalBlockND on MI455X (gfx1250). ~39 GFLOP vs ~134MB HBM -> matrix-core
// bound; all GEMMs via v_wmma_f32_16x16x32_f16 (f16 operands, f32 accum).
// Tile staging uses GLOBAL_LOAD_ASYNC_TO_LDS_B128 (ASYNCcnt) — the CDNA5
// async path — instead of load->ds_store through VGPRs.
// ---------------------------------------------------------------------------

typedef __attribute__((ext_vector_type(16))) _Float16 v16h;
typedef __attribute__((ext_vector_type(8)))  _Float16 v8h;
typedef __attribute__((ext_vector_type(8)))  float    v8f;

#define BATCH 8
#define CCH   512
#define NPOS  4096      // views*t*v
#define MPOS  1024      // views*(t/2)*(v/2)
#define CIN   256
#define CI    128       // C_INTER

// GEMM tiling: 256 threads = 8 waves (wave32), wave grid 2x4.
// Each wave computes a 32x64 C tile = 2x4 WMMA 16x16 frags -> 8 wmma/K-step.
#define BM 64
#define BN 256
#define BK 32
#define LDA 40          // halves, +16B pad
#define LDB 40

#define MODE_THETA 1    // f16 out, transposed:  out[n][m] = acc + bias[m]
#define MODE_CONVH 2    // f16 out, row-major:   out[m][n] = acc + bias[m]
#define MODE_F32   3    // f32 out, row-major:   out[m][n] = acc
#define MODE_Y     4    // f16 out, deinterleave cols: out[(n&1)][m][n>>1] = acc
#define MODE_W     5    // f32 out: out[2m+z][n] = acc + bias[m] + resid[2m+z][n]

__device__ __forceinline__ void async_b128(unsigned lds_off, unsigned long long gaddr) {
  // GLOBAL_LOAD_ASYNC_TO_LDS_B128: VDST = per-lane LDS byte offset, VADDR = 64b
  // global address. Tracked by ASYNCcnt; no VGPR return data.
  asm volatile("global_load_async_to_lds_b128 %0, %1, off"
               :: "v"(lds_off), "v"(gaddr) : "memory");
}
__device__ __forceinline__ void wait_async0() {
  asm volatile("s_wait_asynccnt 0x0" ::: "memory");
}

__launch_bounds__(256)
__global__ void wmma_gemm(const _Float16* __restrict__ A,
                          const _Float16* __restrict__ Bt,
                          const float* __restrict__ bias,
                          const float* __restrict__ resid,
                          void* __restrict__ out,
                          int M, int N, int K,
                          long long strideA, long long strideB, long long strideO,
                          int mode)
{
  __shared__ _Float16 lA[BM * LDA] __attribute__((aligned(32)));
  __shared__ _Float16 lB[BN * LDB] __attribute__((aligned(32)));

  const int tid  = threadIdx.x;
  const int wave = tid >> 5;
  const int lane = tid & 31;
  const int wm   = wave >> 2;      // 0..1
  const int wn   = wave & 3;       // 0..3
  const int lrow = lane & 15;
  const int hi   = lane >> 4;

  const int z = blockIdx.z;
  A  += (long long)z * strideA;
  Bt += (long long)z * strideB;

  const int bm0 = blockIdx.y * BM;
  const int bn0 = blockIdx.x * BN;

  // cooperative tile staging:
  //   A tile 64 rows x 32 halves: thread -> (row=tid>>2, 8-half chunk)
  //   B tile 256 rows x 32 halves: thread -> full row (4 x 16B chunks)
  const int arow = tid >> 2;
  const int acol = (tid & 3) * 8;

  const _Float16* gA = A  + (long long)(bm0 + arow) * K + acol;
  const _Float16* gB = Bt + (long long)(bn0 + tid) * K;

  // flat shared ptr low 32 bits == LDS byte offset (aperture is in high 32b)
  const unsigned ldsA = (unsigned)(unsigned long long)(uintptr_t)&lA[arow * LDA + acol];
  const unsigned ldsB = (unsigned)(unsigned long long)(uintptr_t)&lB[tid * LDB];

  v8f acc[2][4];
  for (int i = 0; i < 2; ++i)
    for (int j = 0; j < 4; ++j)
      for (int e = 0; e < 8; ++e) acc[i][j][e] = 0.0f;

  for (int k0 = 0; k0 < K; k0 += BK) {
    async_b128(ldsA, (unsigned long long)(uintptr_t)(gA + k0));
    #pragma unroll
    for (int j = 0; j < 4; ++j)
      async_b128(ldsB + 16u * j, (unsigned long long)(uintptr_t)(gB + k0 + 8 * j));
    if (k0 + BK < K) {               // global_prefetch_b8 of next K tile
      __builtin_prefetch(gA + k0 + BK, 0, 1);
      __builtin_prefetch(gB + k0 + BK, 0, 1);
    }
    wait_async0();
    __syncthreads();

    v16h af[2], bf[4];
    // A frag (ISA 16-bit A 16x32): lane m=lrow; e<8 -> K=8*hi+e, e>=8 -> 8*hi+16+(e-8)
    #pragma unroll
    for (int mi = 0; mi < 2; ++mi) {
      const int r = wm * 32 + mi * 16 + lrow;
      v8h lo = *(const v8h*)&lA[r * LDA + hi * 8];
      v8h hg = *(const v8h*)&lA[r * LDA + hi * 8 + 16];
      af[mi] = __builtin_shufflevector(lo, hg, 0,1,2,3,4,5,6,7,8,9,10,11,12,13,14,15);
    }
    // B frag (ISA B KxN): lane n=lrow; e<8 -> K=8*hi+e, e>=8 -> 8*hi+16+(e-8)
    #pragma unroll
    for (int ni = 0; ni < 4; ++ni) {
      const int c = wn * 64 + ni * 16 + lrow;
      v8h lo = *(const v8h*)&lB[c * LDB + hi * 8];
      v8h hg = *(const v8h*)&lB[c * LDB + hi * 8 + 16];
      bf[ni] = __builtin_shufflevector(lo, hg, 0,1,2,3,4,5,6,7,8,9,10,11,12,13,14,15);
    }
    #pragma unroll
    for (int mi = 0; mi < 2; ++mi)
      #pragma unroll
      for (int ni = 0; ni < 4; ++ni)
        acc[mi][ni] = __builtin_amdgcn_wmma_f32_16x16x32_f16(
            false, af[mi], false, bf[ni], (short)0, acc[mi][ni], false, false);
    __syncthreads();
  }

  // epilogue: C/D layout — VGPR r, lanes 0-15: (M=r, N=lane); lanes 16-31: (M=8+r)
  #pragma unroll
  for (int mi = 0; mi < 2; ++mi) {
    #pragma unroll
    for (int ni = 0; ni < 4; ++ni) {
      const int rowb = bm0 + wm * 32 + mi * 16 + hi * 8;
      const int ccol = bn0 + wn * 64 + ni * 16 + lrow;
      #pragma unroll
      for (int r = 0; r < 8; ++r) {
        const int crow = rowb + r;
        float v = acc[mi][ni][r];
        if (mode == MODE_THETA) {
          v += bias[crow];
          ((_Float16*)out)[(long long)z * strideO + (long long)ccol * M + crow] = (_Float16)v;
        } else if (mode == MODE_CONVH) {
          v += bias[crow];
          ((_Float16*)out)[(long long)z * strideO + (long long)crow * N + ccol] = (_Float16)v;
        } else if (mode == MODE_F32) {
          ((float*)out)[(long long)z * strideO + (long long)crow * N + ccol] = v;
        } else if (mode == MODE_Y) {
          ((_Float16*)out)[(long long)(ccol & 1) * strideO + (long long)crow * CI + (ccol >> 1)] = (_Float16)v;
        } else { // MODE_W: channel = 2*o + g(z), plus bias and residual x
          const int ch = 2 * crow + z;
          const long long o = (long long)ch * N + ccol;
          ((float*)out)[o] = v + bias[crow] + resid[o];
        }
      }
    }
  }
}

// f32 -> f16 weight convert
__global__ void cvt_f16(const float* __restrict__ in, _Float16* __restrict__ o, int n) {
  int i = blockIdx.x * blockDim.x + threadIdx.x;
  if (i < n) o[i] = (_Float16)in[i];
}

// Stage x: split channels c512 = cin*2 + g into xg[b][g][n][cin] (f16, K-contig)
// and xpool[b][n][cin] = mean over g (f16). LDS transpose keeps both sides coalesced.
__global__ void pack_x(const float* __restrict__ x,
                       _Float16* __restrict__ xg, _Float16* __restrict__ xpool)
{
  __shared__ _Float16 t[256][66];
  const int b  = blockIdx.y;
  const int n0 = blockIdx.x * 64;
  const float* xb = x + (size_t)b * CCH * NPOS;
  for (int h = 0; h < 2; ++h) {
    for (int it = 0; it < 64; ++it) {
      int idx = it * 256 + threadIdx.x;
      int r = idx >> 6, nl = idx & 63;
      t[r][nl] = (_Float16)xb[(size_t)(256 * h + r) * NPOS + n0 + nl];
    }
    __syncthreads();
    for (int it = 0; it < 32; ++it) {
      int idx = it * 256 + threadIdx.x;
      int cl = idx & 127, nl = idx >> 7;
      _Float16 v0 = t[2 * cl][nl];
      _Float16 v1 = t[2 * cl + 1][nl];
      int c = 128 * h + cl, n = n0 + nl;
      xg[((size_t)(b * 2 + 0) * NPOS + n) * CIN + c] = v0;
      xg[((size_t)(b * 2 + 1) * NPOS + n) * CIN + c] = v1;
      xpool[((size_t)b * NPOS + n) * CIN + c] = (_Float16)(0.5f * ((float)v0 + (float)v1));
    }
    __syncthreads();
  }
}

__device__ __forceinline__ int pool_n00(int mp) {
  int view = mp >> 8, rem = mp & 255, th = rem >> 4, vh = rem & 15;
  return view * 1024 + th * 64 + vh * 2;   // n = view*1024 + t*32 + v
}

// phi_full[b][i][n] -> phiX[b][mp][i]  (Bt operand of f-GEMM, K=CI contiguous)
__global__ void pool_phi(const _Float16* __restrict__ pf, _Float16* __restrict__ px) {
  int idx = blockIdx.x * 256 + threadIdx.x;
  int i = idx & 127, mp = (idx >> 7) & 1023, b = idx >> 17;
  const _Float16* s = pf + ((size_t)b * CI + i) * NPOS + pool_n00(mp);
  float m = fmaxf(fmaxf((float)s[0], (float)s[1]), fmaxf((float)s[32], (float)s[33]));
  px[((size_t)b * MPOS + mp) * CI + i] = (_Float16)m;
}

// g_full[b][g][i][n] -> gX[b][d=2i+g][mp]  (Bt operand of y-GEMM, K=MPOS contiguous)
__global__ void pool_g(const _Float16* __restrict__ gf, _Float16* __restrict__ gx) {
  int idx = blockIdx.x * 256 + threadIdx.x;
  int mp = idx & 1023, d = (idx >> 10) & 255, b = idx >> 18;
  int i = d >> 1, g = d & 1;
  const _Float16* s = gf + ((size_t)(b * 2 + g) * CI + i) * NPOS + pool_n00(mp);
  float m = fmaxf(fmaxf((float)s[0], (float)s[1]), fmaxf((float)s[32], (float)s[33]));
  gx[((size_t)b * CIN + d) * MPOS + mp] = (_Float16)m;
}

// softmax over the last dim of f (one row per workgroup), f16 output
__global__ void softmax_rows(const float* __restrict__ f, _Float16* __restrict__ p, int cols)
{
  const int row = blockIdx.x;
  const float* fr = f + (size_t)row * cols;
  const int tid = threadIdx.x;
  __shared__ float red[8];
  __shared__ float sval;

  float lmax = -3.4e38f;
  for (int c = tid; c < cols; c += 256) lmax = fmaxf(lmax, fr[c]);
  #pragma unroll
  for (int off = 16; off; off >>= 1) lmax = fmaxf(lmax, __shfl_xor(lmax, off, 32));
  if ((tid & 31) == 0) red[tid >> 5] = lmax;
  __syncthreads();
  if (tid == 0) {
    float m = red[0];
    for (int i = 1; i < 8; ++i) m = fmaxf(m, red[i]);
    sval = m;
  }
  __syncthreads();
  const float m = sval;
  float lsum = 0.f;
  for (int c = tid; c < cols; c += 256) lsum += __expf(fr[c] - m);
  #pragma unroll
  for (int off = 16; off; off >>= 1) lsum += __shfl_xor(lsum, off, 32);
  __syncthreads();
  if ((tid & 31) == 0) red[tid >> 5] = lsum;
  __syncthreads();
  if (tid == 0) {
    float s = 0.f;
    for (int i = 0; i < 8; ++i) s += red[i];
    sval = 1.f / s;
  }
  __syncthreads();
  const float inv = sval;
  for (int c = tid; c < cols; c += 256)
    p[(size_t)row * cols + c] = (_Float16)(__expf(fr[c] - m) * inv);
}

extern "C" void kernel_launch(void* const* d_in, const int* in_sizes, int n_in,
                              void* d_out, int out_size, void* d_ws, size_t ws_size,
                              hipStream_t stream) {
  (void)in_sizes; (void)n_in; (void)out_size; (void)ws_size;
  const float* x       = (const float*)d_in[0];
  const float* g_w     = (const float*)d_in[1];
  const float* g_b     = (const float*)d_in[2];
  const float* theta_w = (const float*)d_in[3];
  const float* theta_b = (const float*)d_in[4];
  const float* phi_w   = (const float*)d_in[5];
  const float* phi_b   = (const float*)d_in[6];
  const float* W_w     = (const float*)d_in[7];
  const float* W_b     = (const float*)d_in[8];
  float* out = (float*)d_out;

  char* ws = (char*)d_ws;
  auto alloc = [&](size_t bytes) {
    char* p = ws;
    ws += (bytes + 255) & ~(size_t)255;
    return p;
  };
  _Float16* wG     = (_Float16*)alloc((size_t)CI * CIN * 2);
  _Float16* wTheta = (_Float16*)alloc((size_t)CI * CIN * 2);
  _Float16* wPhi   = (_Float16*)alloc((size_t)CI * CIN * 2);
  _Float16* wW     = (_Float16*)alloc((size_t)CIN * CI * 2);
  _Float16* xpool  = (_Float16*)alloc((size_t)BATCH * NPOS * CIN * 2);
  _Float16* xg     = (_Float16*)alloc((size_t)BATCH * 2 * NPOS * CIN * 2);
  _Float16* thetaX = (_Float16*)alloc((size_t)BATCH * NPOS * CI * 2);   // [b][n][i]
  _Float16* phiF   = (_Float16*)alloc((size_t)BATCH * CI * NPOS * 2);   // [b][i][n]
  _Float16* phiX   = (_Float16*)alloc((size_t)BATCH * MPOS * CI * 2);   // [b][mp][i]
  _Float16* gF     = (_Float16*)alloc((size_t)BATCH * 2 * CI * NPOS * 2);
  _Float16* gX     = (_Float16*)alloc((size_t)BATCH * CIN * MPOS * 2);  // [b][d][mp]
  float*    fbuf   = (float*)   alloc((size_t)NPOS * MPOS * 4);         // per-batch reuse
  _Float16* ph     = (_Float16*)alloc((size_t)NPOS * MPOS * 2);         // per-batch reuse
  _Float16* yT     = (_Float16*)alloc((size_t)2 * NPOS * CI * 2);       // [g][n][i], reuse

  // weights -> f16
  cvt_f16<<<128, 256, 0, stream>>>(g_w,     wG,     CI * CIN);
  cvt_f16<<<128, 256, 0, stream>>>(theta_w, wTheta, CI * CIN);
  cvt_f16<<<128, 256, 0, stream>>>(phi_w,   wPhi,   CI * CIN);
  cvt_f16<<<128, 256, 0, stream>>>(W_w,     wW,     CIN * CI);

  // stage x
  pack_x<<<dim3(NPOS / 64, BATCH), 256, 0, stream>>>(x, xg, xpool);

  const long long sBx = (long long)NPOS * CIN;
  // theta: (128x256)x(256x4096) per batch -> thetaX[b][n][i]
  wmma_gemm<<<dim3(NPOS / BN, CI / BM, BATCH), 256, 0, stream>>>(
      wTheta, xpool, theta_b, nullptr, thetaX,
      CI, NPOS, CIN, 0LL, sBx, (long long)NPOS * CI, MODE_THETA);
  // phi: same, row-major f16 full-res -> pooled later
  wmma_gemm<<<dim3(NPOS / BN, CI / BM, BATCH), 256, 0, stream>>>(
      wPhi, xpool, phi_b, nullptr, phiF,
      CI, NPOS, CIN, 0LL, sBx, (long long)CI * NPOS, MODE_CONVH);
  // g: per (b,g)
  wmma_gemm<<<dim3(NPOS / BN, CI / BM, 2 * BATCH), 256, 0, stream>>>(
      wG, xg, g_b, nullptr, gF,
      CI, NPOS, CIN, 0LL, sBx, (long long)CI * NPOS, MODE_CONVH);

  pool_phi<<<(BATCH * MPOS * CI) / 256, 256, 0, stream>>>(phiF, phiX);
  pool_g<<<(BATCH * CIN * MPOS) / 256, 256, 0, stream>>>(gF, gX);

  for (int b = 0; b < BATCH; ++b) {
    // f = theta_x @ phi_x : (4096x128)x(128x1024) -> f32
    wmma_gemm<<<dim3(MPOS / BN, NPOS / BM, 1), 256, 0, stream>>>(
        thetaX + (size_t)b * NPOS * CI, phiX + (size_t)b * MPOS * CI,
        nullptr, nullptr, fbuf, NPOS, MPOS, CI, 0LL, 0LL, 0LL, MODE_F32);
    // softmax over M
    softmax_rows<<<NPOS, 256, 0, stream>>>(fbuf, ph, MPOS);
    // y = p @ g_x : (4096x1024)x(1024x256) -> yT[g][n][i]
    wmma_gemm<<<dim3(CIN / BN, NPOS / BM, 1), 256, 0, stream>>>(
        ph, gX + (size_t)b * CIN * MPOS, nullptr, nullptr, yT,
        NPOS, CIN, MPOS, 0LL, 0LL, (long long)NPOS * CI, MODE_Y);
    // W conv + bias + residual: per g, (256x128)x(128x4096) -> out[b][2o+g][n]
    wmma_gemm<<<dim3(NPOS / BN, CIN / BM, 2), 256, 0, stream>>>(
        wW, yT, W_b, x + (size_t)b * CCH * NPOS, out + (size_t)b * CCH * NPOS,
        CIN, NPOS, CI, 0LL, (long long)NPOS * CI, 0LL, MODE_W);
  }
}
